// SAGE_57105885167813
// MI455X (gfx1250) — matrix operand order repaired and checked by reading, not compiled, not run on hardware
//
#include <hip/hip_runtime.h>

// ---------------------------------------------------------------------------
// GraphSAGE (3x SAGEConv + BN + ReLU, concat, MLP 512->128->64) for gfx1250.
// Aggregation is L2-resident scatter/gather (192MB L2 >> 65MB working set).
// All GEMMs run on v_wmma_f32_16x16x32_bf16 with bf16 hi/lo error-compensated
// split (3 WMMA per tile ~ fp32 accuracy). BN stats fused into GEMM epilogue.
// ---------------------------------------------------------------------------

typedef unsigned short u16;
typedef __attribute__((ext_vector_type(16))) __bf16 v16bf;
typedef __attribute__((ext_vector_type(8)))  float  v8f;

union BF16Frag {
  v16bf v;
  u16   s[16];
  uint4 q[2];
};

__device__ __forceinline__ u16 f2bf(float x) {
  unsigned u = __builtin_bit_cast(unsigned, x);
  unsigned r = u + 0x7FFFu + ((u >> 16) & 1u);   // round-to-nearest-even
  return (u16)(r >> 16);
}
__device__ __forceinline__ float bf2f(u16 b) {
  unsigned u = ((unsigned)b) << 16;
  return __builtin_bit_cast(float, u);
}

// ---------------------------------------------------------------------------
// Degree / inverse degree
// ---------------------------------------------------------------------------
__global__ void degree_kernel(const int* __restrict__ dst, float* __restrict__ deg, int nE) {
  int i = blockIdx.x * blockDim.x + threadIdx.x;
  if (i < nE) atomicAdd(&deg[dst[i]], 1.0f);
}

__global__ void invdeg_kernel(float* __restrict__ deg, int n) {
  int i = blockIdx.x * blockDim.x + threadIdx.x;
  if (i < n) deg[i] = 1.0f / fmaxf(deg[i], 1.0f);
}

// ---------------------------------------------------------------------------
// Copy x (N x 128) into collect[:, 0:128] (pitch 512)
// ---------------------------------------------------------------------------
__global__ void copy_x_kernel(const float* __restrict__ x, float* __restrict__ collect, int total) {
  int i = blockIdx.x * blockDim.x + threadIdx.x;
  if (i >= total) return;
  int m = i >> 7, c = i & 127;
  collect[(size_t)m * 512 + c] = x[i];
}

// ---------------------------------------------------------------------------
// Edge scatter-add: agg[dst] += h[src]  (one wave per edge, float4 per lane)
// h rows live in L2; prefetch the next edge's source row to hide latency.
// ---------------------------------------------------------------------------
__global__ __launch_bounds__(256) void scatter_add_kernel(
    const int* __restrict__ src, const int* __restrict__ dst,
    const float* __restrict__ h, int ldh,
    float* __restrict__ agg, int nE)
{
  int lane = threadIdx.x & 31;
  int wave = (blockIdx.x * blockDim.x + threadIdx.x) >> 5;
  int nw   = (gridDim.x * blockDim.x) >> 5;
  for (int e = wave; e < nE; e += nw) {
    int s = src[e];
    int d = dst[e];
    if (e + nw < nE) {
      int s2 = src[e + nw];
      __builtin_prefetch(h + (size_t)s2 * ldh + lane * 4, 0, 1);  // global_prefetch
    }
    float4 v = *(const float4*)(h + (size_t)s * ldh + lane * 4);
    float* o = agg + (size_t)d * 128 + lane * 4;
    atomicAdd(o + 0, v.x);
    atomicAdd(o + 1, v.y);
    atomicAdd(o + 2, v.z);
    atomicAdd(o + 3, v.w);
  }
}

// ---------------------------------------------------------------------------
// Pack f32 weight [K x Nout] into WMMA B-fragment order, bf16 hi/lo split.
// Fragment layout (per 32x16 K-tile x N-tile): lane l elem e holds B[kbase + (e<8 ? (l/16)*8+e : 16+(l/16)*8+e-8)][nbase + l%16].
// ---------------------------------------------------------------------------
__global__ void pack_weight_kernel(const float* __restrict__ W,
                                   u16* __restrict__ hi, u16* __restrict__ lo,
                                   int K, int Nout)
{
  int idx = blockIdx.x * blockDim.x + threadIdx.x;
  if (idx >= K * Nout) return;
  int k = idx / Nout, n = idx % Nout;
  int kt = k >> 5, kk = k & 31;
  int half = (kk >> 3) & 1, e = (kk & 7) + ((kk >> 4) << 3);
  int nt = n >> 4, nn = n & 15;
  int nNt = Nout >> 4;
  size_t p = (((size_t)(kt * nNt + nt) * 32) + half * 16 + nn) * 16 + e;
  float x = W[idx];
  u16 hb = f2bf(x);
  hi[p] = hb;
  lo[p] = f2bf(x - bf2f(hb));
}

// ---------------------------------------------------------------------------
// One K-pass of the WMMA GEMM: acc += split_bf16(A*rs) @ split_bf16(B)
// A fragment per lane: row m = lane%16; e in [0,8) -> k = half*8+e,
// e in [8,16) -> k = 16 + half*8 + (e-8)  (documented 16-bit A 16x32 layout).
// ---------------------------------------------------------------------------
template<int NT>
__device__ __forceinline__ void gemm_pass(v8f (&acc)[NT],
    const float* __restrict__ A, int lda, float rs,
    const u16* __restrict__ Bhi, const u16* __restrict__ Blo,
    int ktiles, int rowA, int half, int lane)
{
  for (int kt = 0; kt < ktiles; ++kt) {
    const float* ar = A + (size_t)rowA * lda + kt * 32 + half * 8;
    float4 f0 = *(const float4*)(ar + 0);
    float4 f1 = *(const float4*)(ar + 4);
    float4 f2 = *(const float4*)(ar + 16);
    float4 f3 = *(const float4*)(ar + 20);
    float av[16] = {f0.x, f0.y, f0.z, f0.w, f1.x, f1.y, f1.z, f1.w,
                    f2.x, f2.y, f2.z, f2.w, f3.x, f3.y, f3.z, f3.w};
    BF16Frag ahi, alo;
#pragma unroll
    for (int e = 0; e < 16; ++e) {
      float xv = av[e] * rs;
      u16 hb = f2bf(xv);
      ahi.s[e] = hb;
      alo.s[e] = f2bf(xv - bf2f(hb));
    }
#pragma unroll
    for (int t = 0; t < NT; ++t) {
      size_t base = ((size_t)(kt * NT + t) * 32 + lane) * 16;
      const uint4* ph = (const uint4*)(Bhi + base);
      const uint4* pl = (const uint4*)(Blo + base);
      BF16Frag bhi, blo;
      bhi.q[0] = ph[0]; bhi.q[1] = ph[1];
      blo.q[0] = pl[0]; blo.q[1] = pl[1];
      // bf16x3 compensated product: hi*hi + hi*lo + lo*hi  (~fp32 accuracy)
      acc[t] = __builtin_amdgcn_wmma_f32_16x16x32_bf16(false, ahi.v, false, bhi.v, (short)0, acc[t], false, false);
      acc[t] = __builtin_amdgcn_wmma_f32_16x16x32_bf16(false, ahi.v, false, blo.v, (short)0, acc[t], false, false);
      acc[t] = __builtin_amdgcn_wmma_f32_16x16x32_bf16(false, alo.v, false, bhi.v, (short)0, acc[t], false, false);
    }
  }
}

// ---------------------------------------------------------------------------
// WMMA GEMM: out[M x NT*16] = diag(scale0)*A0 @ B0 (+ A1 @ B1) (+ bias)
// One wave computes 16 rows x NT*16 cols. Fused BN column-stat reduction.
// ---------------------------------------------------------------------------
template<int NT, bool HAS2, bool HAS_BIAS, bool HAS_STATS, bool SCALE0>
__global__ __launch_bounds__(256) void gemm_bf16x3_kernel(
    const float* __restrict__ A0, int lda0, const float* __restrict__ scale0,
    const u16* __restrict__ B0hi, const u16* __restrict__ B0lo, int kt0,
    const float* __restrict__ A1, int lda1,
    const u16* __restrict__ B1hi, const u16* __restrict__ B1lo, int kt1,
    const float* __restrict__ bias,
    float* __restrict__ out, int ldo,
    float* __restrict__ statSum, float* __restrict__ statSq,
    int M)
{
  const int lane = threadIdx.x & 31;
  const int tile = blockIdx.x * (blockDim.x >> 5) + (threadIdx.x >> 5);
  const int m0 = tile * 16;
  if (m0 >= M) return;                 // whole-wave exit: EXEC stays all-ones
  const int half = lane >> 4;
  const int n    = lane & 15;
  const int rowA = m0 + n;             // lane's A-matrix row (m = lane%16)

  v8f zero = {0.f, 0.f, 0.f, 0.f, 0.f, 0.f, 0.f, 0.f};
  v8f acc[NT];
#pragma unroll
  for (int t = 0; t < NT; ++t) acc[t] = zero;

  float rs = SCALE0 ? scale0[rowA] : 1.0f;
  gemm_pass<NT>(acc, A0, lda0, rs, B0hi, B0lo, kt0, rowA, half, lane);
  if (HAS2)
    gemm_pass<NT>(acc, A1, lda1, 1.0f, B1hi, B1lo, kt1, rowA, half, lane);

  if (HAS_BIAS) {
#pragma unroll
    for (int t = 0; t < NT; ++t) {
      float bv = bias[t * 16 + n];
#pragma unroll
      for (int r = 0; r < 8; ++r) acc[t][r] += bv;
    }
  }

#pragma unroll
  for (int t = 0; t < NT; ++t) {
    float cs = 0.f, cq = 0.f;
#pragma unroll
    for (int r = 0; r < 8; ++r) {
      float v = acc[t][r];                             // C[m0 + half*8 + r][t*16+n]
      out[(size_t)(m0 + half * 8 + r) * ldo + t * 16 + n] = v;
      if (HAS_STATS) { cs += v; cq += v * v; }
    }
    if (HAS_STATS) {
      cs += __shfl_xor(cs, 16, 32);                    // combine row halves (wave32)
      cq += __shfl_xor(cq, 16, 32);
      if (half == 0) {
        atomicAdd(&statSum[t * 16 + n], cs);
        atomicAdd(&statSq [t * 16 + n], cq);
      }
    }
  }
}

// ---------------------------------------------------------------------------
// BN: fold stats into per-column scale/shift:  y = v*a + b
// ---------------------------------------------------------------------------
__global__ void bn_finalize_kernel(const float* __restrict__ sum, const float* __restrict__ sq,
                                   const float* __restrict__ gamma, const float* __restrict__ beta,
                                   float* __restrict__ coefA, float* __restrict__ coefB,
                                   int C, float invM)
{
  int c = blockIdx.x * blockDim.x + threadIdx.x;
  if (c >= C) return;
  float mu  = sum[c] * invM;
  float var = sq[c] * invM - mu * mu;
  float a   = gamma[c] * rsqrtf(var + 1e-5f);
  coefA[c] = a;
  coefB[c] = beta[c] - mu * a;
}

// in-place BN + (leaky-)ReLU over a [M x C] region with pitch ld
__global__ void bn_apply_kernel(float* __restrict__ buf, int ld,
                                const float* __restrict__ coefA, const float* __restrict__ coefB,
                                int M, int C, float slope)
{
  int i = blockIdx.x * blockDim.x + threadIdx.x;
  if (i >= M * C) return;
  int m = i / C, c = i % C;
  size_t o = (size_t)m * ld + c;
  float v = buf[o] * coefA[c] + coefB[c];
  buf[o] = v > 0.f ? v : slope * v;
}

// ---------------------------------------------------------------------------
// Host orchestration
// ---------------------------------------------------------------------------
extern "C" void kernel_launch(void* const* d_in, const int* in_sizes, int n_in,
                              void* d_out, int out_size, void* d_ws, size_t ws_size,
                              hipStream_t stream)
{
  (void)n_in; (void)out_size; (void)ws_size;
  const int N = 50000;

  const int*   edges = (const int*)d_in[0];
  const int    nE    = in_sizes[0] / 2;
  const int*   src   = edges;
  const int*   dst   = edges + nE;
  const float* x     = (const float*)d_in[1];
  const float* Wl    = (const float*)d_in[2];   // [3][128][128]  (K x Nout)
  const float* Wr    = (const float*)d_in[3];
  const float* gamma = (const float*)d_in[4];
  const float* beta  = (const float*)d_in[5];
  const float* W1    = (const float*)d_in[6];   // [512][128]
  const float* b1    = (const float*)d_in[7];
  const float* g1    = (const float*)d_in[8];
  const float* be1   = (const float*)d_in[9];
  const float* W2    = (const float*)d_in[10];  // [128][64]
  const float* b2    = (const float*)d_in[11];
  float*       outp  = (float*)d_out;           // [N][64]

  // ---- workspace carve-up (~130 MB) ----
  char*  ws  = (char*)d_ws;
  size_t off = 0;
  auto alloc = [&](size_t bytes) -> char* {
    char* p = ws + off;
    off += (bytes + 255) & ~(size_t)255;
    return p;
  };
  float* collect = (float*)alloc((size_t)N * 512 * 4);  // [x | h1 | h2 | h3]
  float* agg     = (float*)alloc((size_t)N * 128 * 4);  // aggregation / MLP hidden
  float* invdeg  = (float*)alloc((size_t)N * 4);
  float* statSum = (float*)alloc(128 * 4);
  float* statSq  = (float*)alloc(128 * 4);
  float* coefA   = (float*)alloc(128 * 4);
  float* coefB   = (float*)alloc(128 * 4);
  u16* pWl_hi[3], *pWl_lo[3], *pWr_hi[3], *pWr_lo[3];
  for (int i = 0; i < 3; ++i) {
    pWl_hi[i] = (u16*)alloc(128 * 128 * 2);  pWl_lo[i] = (u16*)alloc(128 * 128 * 2);
    pWr_hi[i] = (u16*)alloc(128 * 128 * 2);  pWr_lo[i] = (u16*)alloc(128 * 128 * 2);
  }
  u16* pW1_hi = (u16*)alloc(512 * 128 * 2);
  u16* pW1_lo = (u16*)alloc(512 * 128 * 2);
  u16* pW2_hi = (u16*)alloc(128 * 64 * 2);
  u16* pW2_lo = (u16*)alloc(128 * 64 * 2);

  const int TILES       = N / 16;                 // 3125 (exact)
  const int GEMM_BLOCKS = (TILES + 7) / 8;        // 8 waves per block
  const int EW_BLOCKS   = (N * 128 + 255) / 256;  // elementwise over N x 128

  // ---- degree ----
  hipMemsetAsync(invdeg, 0, (size_t)N * 4, stream);
  degree_kernel<<<(nE + 255) / 256, 256, 0, stream>>>(dst, invdeg, nE);
  invdeg_kernel<<<(N + 255) / 256, 256, 0, stream>>>(invdeg, N);

  // ---- collect[:,0:128] = x ----
  copy_x_kernel<<<EW_BLOCKS, 256, 0, stream>>>(x, collect, N * 128);

  // ---- pack all weights into WMMA B-fragment order (bf16 hi/lo) ----
  for (int i = 0; i < 3; ++i) {
    pack_weight_kernel<<<(128 * 128 + 255) / 256, 256, 0, stream>>>(Wl + (size_t)i * 128 * 128, pWl_hi[i], pWl_lo[i], 128, 128);
    pack_weight_kernel<<<(128 * 128 + 255) / 256, 256, 0, stream>>>(Wr + (size_t)i * 128 * 128, pWr_hi[i], pWr_lo[i], 128, 128);
  }
  pack_weight_kernel<<<(512 * 128 + 255) / 256, 256, 0, stream>>>(W1, pW1_hi, pW1_lo, 512, 128);
  pack_weight_kernel<<<(128 * 64 + 255) / 256, 256, 0, stream>>>(W2, pW2_hi, pW2_lo, 128, 64);

  // ---- 3 SAGE layers ----
  for (int i = 0; i < 3; ++i) {
    const float* h = collect + 128 * i;           // current features, pitch 512
    float* hnext   = collect + 128 * (i + 1);

    hipMemsetAsync(agg, 0, (size_t)N * 128 * 4, stream);
    scatter_add_kernel<<<(nE + 7) / 8, 256, 0, stream>>>(src, dst, h, 512, agg, nE);

    hipMemsetAsync(statSum, 0, 128 * 4, stream);
    hipMemsetAsync(statSq,  0, 128 * 4, stream);
    // hnext_pre = (agg * invdeg) @ Wl + h @ Wr   (fused BN stats)
    gemm_bf16x3_kernel<8, true, false, true, true><<<GEMM_BLOCKS, 256, 0, stream>>>(
        agg, 128, invdeg, pWl_hi[i], pWl_lo[i], 4,
        h, 512, pWr_hi[i], pWr_lo[i], 4,
        nullptr, hnext, 512, statSum, statSq, N);

    bn_finalize_kernel<<<1, 128, 0, stream>>>(statSum, statSq, gamma + i * 128, beta + i * 128,
                                              coefA, coefB, 128, 1.0f / N);
    bn_apply_kernel<<<EW_BLOCKS, 256, 0, stream>>>(hnext, 512, coefA, coefB, N, 128, 0.0f);
  }

  // ---- MLP layer 1: z = collect @ W1 + b1 ; BN ; leaky_relu(0.01) ----
  hipMemsetAsync(statSum, 0, 128 * 4, stream);
  hipMemsetAsync(statSq,  0, 128 * 4, stream);
  gemm_bf16x3_kernel<8, false, true, true, false><<<GEMM_BLOCKS, 256, 0, stream>>>(
      collect, 512, nullptr, pW1_hi, pW1_lo, 16,
      nullptr, 0, nullptr, nullptr, 0,
      b1, agg, 128, statSum, statSq, N);
  bn_finalize_kernel<<<1, 128, 0, stream>>>(statSum, statSq, g1, be1, coefA, coefB, 128, 1.0f / N);
  bn_apply_kernel<<<EW_BLOCKS, 256, 0, stream>>>(agg, 128, coefA, coefB, N, 128, 0.01f);

  // ---- MLP layer 2: out = z @ W2 + b2 ----
  gemm_bf16x3_kernel<4, false, true, false, false><<<GEMM_BLOCKS, 256, 0, stream>>>(
      agg, 128, nullptr, pW2_hi, pW2_lo, 4,
      nullptr, 0, nullptr, nullptr, 0,
      b2, outp, 64, nullptr, nullptr, N);
}
